// GCNEncoder_87471303950923
// MI455X (gfx1250) — matrix operand order repaired and checked by reading, not compiled, not run on hardware
//
#include <hip/hip_runtime.h>

#define N_NODES 50000
#define N_EDGES 800000
#define CH 128
#define TILES_M (N_NODES / 16)   // 3125, exact
#define TILES_N (CH / 16)        // 8

typedef __bf16 bf16_t;
typedef bf16_t v16bf __attribute__((ext_vector_type(16)));
typedef float  v8f   __attribute__((ext_vector_type(8)));

// ---------------- degree / normalization ----------------

__global__ __launch_bounds__(256) void init_deg(float* __restrict__ deg) {
    int i = blockIdx.x * 256 + threadIdx.x;
    if (i < N_NODES) deg[i] = 1.0f;  // self-loop
}

__global__ __launch_bounds__(256) void accum_deg(const int* __restrict__ dst,
                                                 float* __restrict__ deg) {
    int e = blockIdx.x * 256 + threadIdx.x;
    if (e < N_EDGES) atomicAdd(&deg[dst[e]], 1.0f);
}

__global__ __launch_bounds__(256) void rsqrt_deg(float* __restrict__ deg) {
    int i = blockIdx.x * 256 + threadIdx.x;
    if (i < N_NODES) deg[i] = rsqrtf(deg[i]);
}

// ---------------- weight convert (fp32 row-major [k][n] -> bf16 transposed [n][k]) ----------------

__global__ __launch_bounds__(256) void conv_w_bf16_t(const float* __restrict__ W,
                                                     bf16_t* __restrict__ Wt) {
    int idx = blockIdx.x * 256 + threadIdx.x;   // CH*CH = 16384
    if (idx >= CH * CH) return;
    int n = idx & (CH - 1);
    int k = idx >> 7;
    Wt[n * CH + k] = (bf16_t)W[k * CH + n];
}

// ---------------- WMMA GEMM: Out = [relu](A @ W [+ bias]) ----------------
// A: [N_NODES, CH] fp32 row-major (converted to bf16 on the fly)
// Wt: [CH(n), CH(k)] bf16 (pre-transposed so B K-pairs are contiguous)
// One wave computes one 16x16 output tile; K=128 -> 4 x wmma_f32_16x16x32_bf16.

__global__ __launch_bounds__(256) void gemm_wmma_bf16(const float* __restrict__ A,
                                                      const bf16_t* __restrict__ Wt,
                                                      const float* __restrict__ bias,
                                                      float* __restrict__ Out,
                                                      int applyRelu) {
    int wave = threadIdx.x >> 5;
    int lane = threadIdx.x & 31;
    int tile = blockIdx.x * 8 + wave;
    if (tile >= TILES_M * TILES_N) return;
    int tm = tile / TILES_N;
    int tn = tile % TILES_N;
    int m0 = tm * 16, n0 = tn * 16;
    int half = lane >> 4;      // lane group
    int lidx = lane & 15;
    int m = m0 + lidx;         // A row for this lane
    int n = n0 + lidx;         // B/C/D column for this lane

    const float*  arow = A  + (size_t)m * CH;
    const bf16_t* bcol = Wt + (size_t)n * CH;

    v8f c = {0.f, 0.f, 0.f, 0.f, 0.f, 0.f, 0.f, 0.f};

#pragma unroll
    for (int kc = 0; kc < 4; ++kc) {
        int kb = kc * 32;
        v16bf a, b;
#pragma unroll
        for (int v = 0; v < 8; ++v) {
            // A fragment: lanes 0-15 hold K pairs {0..7,16..23}, lanes 16-31 {8..15,24..31}
            int ka = kb + ((v < 4) ? 0 : 16) + half * 8 + (v & 3) * 2;
            float2 f = *(const float2*)(arow + ka);
            a[2 * v]     = (bf16_t)f.x;
            a[2 * v + 1] = (bf16_t)f.y;
            // B fragment: same K order as A (lanes 0-15 {0..7,16..23}, lanes 16-31 {8..15,24..31})
            int kv = kb + ((v < 4) ? 0 : 16) + half * 8 + (v & 3) * 2;
            b[2 * v]     = bcol[kv];
            b[2 * v + 1] = bcol[kv + 1];
        }
        c = __builtin_amdgcn_wmma_f32_16x16x32_bf16(false, a, false, b,
                                                    (short)0, c, false, false);
    }

    float bv = bias ? bias[n] : 0.0f;
#pragma unroll
    for (int r = 0; r < 8; ++r) {
        int mr = m0 + r + 8 * half;   // C/D layout: VGPR r -> M=r (lanes 0-15), M=8+r (lanes 16-31)
        float v = c[r] + bv;
        if (applyRelu) v = fmaxf(v, 0.0f);
        Out[(size_t)mr * CH + n] = v;
    }
}

// ---------------- bias broadcast init: out[i, :] = b ----------------

__global__ __launch_bounds__(256) void init_bias(const float* __restrict__ b,
                                                 float* __restrict__ out) {
    int idx = blockIdx.x * 256 + threadIdx.x;   // N_NODES * 32 (float4 each)
    if (idx >= N_NODES * 32) return;
    int i  = idx >> 5;
    int c4 = (idx & 31) * 4;
    float4 bv = *(const float4*)(b + c4);
    *(float4*)(out + (size_t)i * CH + c4) = bv;
}

// ---------------- edge scatter: out[dst] += hw[src] * dis[src]*dis[dst] ----------------
// Self-loops synthesized for e in [N_EDGES, N_EDGES+N_NODES). 4 channels per thread.

__global__ __launch_bounds__(256) void scatter_edges(const int* __restrict__ ei,
                                                     const float* __restrict__ dis,
                                                     const float* __restrict__ hw,
                                                     float* __restrict__ out) {
    const int ETOT = N_EDGES + N_NODES;
    int idx = blockIdx.x * 256 + threadIdx.x;   // ETOT*32 threads
    if (idx >= ETOT * 32) return;
    int e  = idx >> 5;
    int c4 = (idx & 31) * 4;
    int s, d;
    if (e < N_EDGES) {
        s = ei[e];
        d = ei[N_EDGES + e];
    } else {
        s = d = e - N_EDGES;
    }
    float norm = dis[s] * dis[d];
    float4 v = *(const float4*)(hw + (size_t)s * CH + c4);
    float* o = out + (size_t)d * CH + c4;
    atomicAdd(o + 0, v.x * norm);
    atomicAdd(o + 1, v.y * norm);
    atomicAdd(o + 2, v.z * norm);
    atomicAdd(o + 3, v.w * norm);
}

// ---------------- in-place ReLU ----------------

__global__ __launch_bounds__(256) void relu_inplace(float* __restrict__ x) {
    int idx = blockIdx.x * 256 + threadIdx.x;   // N_NODES*32 float4
    if (idx >= N_NODES * 32) return;
    float4* p = (float4*)x + idx;
    float4 v = *p;
    v.x = fmaxf(v.x, 0.f); v.y = fmaxf(v.y, 0.f);
    v.z = fmaxf(v.z, 0.f); v.w = fmaxf(v.w, 0.f);
    *p = v;
}

// ---------------- host orchestration ----------------

extern "C" void kernel_launch(void* const* d_in, const int* in_sizes, int n_in,
                              void* d_out, int out_size, void* d_ws, size_t ws_size,
                              hipStream_t stream) {
    const float* x      = (const float*)d_in[0];
    const int*   ei     = (const int*)  d_in[1];   // [2, N_EDGES]
    const float* lin_w  = (const float*)d_in[2];   // [CH, CH]
    const float* lin_b  = (const float*)d_in[3];   // [CH]
    const float* conv_w = (const float*)d_in[4];   // [2, CH, CH]
    const float* conv_b = (const float*)d_in[5];   // [2, CH]
    float* out = (float*)d_out;                    // [N_NODES, CH]

    // workspace layout (256-byte aligned regions)
    char* ws = (char*)d_ws;
    size_t off = 0;
    float* dis = (float*)(ws + off);  off += ((size_t)N_NODES * 4 + 255) & ~(size_t)255;
    float* hbuf  = (float*)(ws + off); off += (size_t)N_NODES * CH * 4;
    float* hwbuf = (float*)(ws + off); off += (size_t)N_NODES * CH * 4;
    bf16_t* wtbuf = (bf16_t*)(ws + off); off += (size_t)CH * CH * 2;
    (void)ws_size; (void)n_in; (void)in_sizes; (void)out_size;

    const int nB   = (N_NODES + 255) / 256;          // node-wise
    const int eB   = (N_EDGES + 255) / 256;          // edge-wise
    const int n32B = (N_NODES * 32 + 255) / 256;     // node*32 (float4 rows)
    const int e32B = ((N_EDGES + N_NODES) * 32 + 255) / 256;
    const int wB   = (CH * CH + 255) / 256;
    const int gB   = (TILES_M * TILES_N + 7) / 8;    // 8 waves/block

    // degree normalization
    init_deg<<<nB, 256, 0, stream>>>(dis);
    accum_deg<<<eB, 256, 0, stream>>>(ei + N_EDGES, dis);
    rsqrt_deg<<<nB, 256, 0, stream>>>(dis);

    // h0 = relu(x @ lin_w + lin_b)  -> hbuf
    conv_w_bf16_t<<<wB, 256, 0, stream>>>(lin_w, wtbuf);
    gemm_wmma_bf16<<<gB, 256, 0, stream>>>(x, wtbuf, lin_b, hbuf, 1);

    // layer 0: hw = h0 @ W0 ; agg(+bias) -> hbuf ; relu
    conv_w_bf16_t<<<wB, 256, 0, stream>>>(conv_w, wtbuf);
    gemm_wmma_bf16<<<gB, 256, 0, stream>>>(hbuf, wtbuf, nullptr, hwbuf, 0);
    init_bias<<<n32B, 256, 0, stream>>>(conv_b, hbuf);
    scatter_edges<<<e32B, 256, 0, stream>>>(ei, dis, hwbuf, hbuf);
    relu_inplace<<<n32B, 256, 0, stream>>>(hbuf);

    // layer 1: hw = h1 @ W1 ; agg(+bias) -> d_out ; relu
    conv_w_bf16_t<<<wB, 256, 0, stream>>>(conv_w + CH * CH, wtbuf);
    gemm_wmma_bf16<<<gB, 256, 0, stream>>>(hbuf, wtbuf, nullptr, hwbuf, 0);
    init_bias<<<n32B, 256, 0, stream>>>(conv_b + CH, out);
    scatter_edges<<<e32B, 256, 0, stream>>>(ei, dis, hwbuf, out);
    relu_inplace<<<n32B, 256, 0, stream>>>(out);
}
